// HarmonicOscillatorCell_56289841381380
// MI455X (gfx1250) — matrix operand order repaired and checked by reading, not compiled, not run on hardware
//
#include <hip/hip_runtime.h>
#include <hip/hip_bf16.h>

typedef __attribute__((ext_vector_type(16))) __bf16 v16bf;
typedef __attribute__((ext_vector_type(8)))  __bf16 v8bf;
typedef __attribute__((ext_vector_type(8)))  float  v8f;

#define DT_STEP 0.1f
#define BSZ     4
#define TSEQ    4096
#define DDIM    1024
#define CHUNKS  32
#define CLEN    128                      /* TSEQ / CHUNKS */
#define NTOT    ((size_t)BSZ * TSEQ * DDIM)
#define WTOT    ((size_t)DDIM * DDIM)

__device__ __forceinline__ float softplus_f(float w) {
    return (w > 20.0f) ? w : log1pf(expf(w));
}

// Split fp32 into bf16 hi (truncated) + bf16 lo (residual, rounded).
__device__ __forceinline__ void split_bf16(float x, __bf16 &hi, __bf16 &lo) {
    union { float f; unsigned u; } t;
    t.f = x;
    t.u &= 0xffff0000u;                  // exact bf16 value
    hi = (__bf16)t.f;
    lo = (__bf16)(x - t.f);
}

__device__ __forceinline__ v16bf cat8(v8bf a, v8bf b) {
    return __builtin_shufflevector(a, b, 0, 1, 2, 3, 4, 5, 6, 7,
                                         8, 9, 10, 11, 12, 13, 14, 15);
}

// -------------------------------------------------------------------------
// Phase -1: one-pass fp32 -> bf16 hi/lo decomposition (8 elements/thread).
// Done once so the GEMM inner loop carries zero conversion VALU.
// -------------------------------------------------------------------------
__global__ __launch_bounds__(256) void convert_hilo(
        const float* __restrict__ src,
        __bf16* __restrict__ hi, __bf16* __restrict__ lo) {
    const size_t i = (size_t)blockIdx.x * 256 + threadIdx.x;
    const float4* p = (const float4*)src + 2 * i;
    float4 f0 = p[0], f1 = p[1];
    float vals[8] = {f0.x, f0.y, f0.z, f0.w, f1.x, f1.y, f1.z, f1.w};
    v8bf hv, lv;
#pragma unroll
    for (int j = 0; j < 8; ++j) {
        __bf16 h, l;
        split_bf16(vals[j], h, l);
        hv[j] = h;
        lv[j] = l;
    }
    *(v8bf*)(hi + 8 * i) = hv;
    *(v8bf*)(lo + 8 * i) = lv;
}

// -------------------------------------------------------------------------
// Phase 0: Bx = x_seq @ B_weight^T via bf16 hi/lo split WMMA (3 products:
// hi*hi + hi*lo + lo*hi, ~2^-22 relative error vs fp32).
// Each wave computes a 32(M) x 64(N) strip: 8 accumulator tiles; A fragments
// reused across 4 N-tiles, B fragments across 2 M-tiles (24 b128 loads :
// 24 WMMAs per K-chunk).  Block = 128 threads (4 waves).
// grid = (M/32, 4); wave w covers N-group blockIdx.y*4+w (64 columns).
// -------------------------------------------------------------------------
__global__ __launch_bounds__(128) void gemm_bx_kernel(
        const __bf16* __restrict__ Xhi, const __bf16* __restrict__ Xlo,
        const __bf16* __restrict__ Whi, const __bf16* __restrict__ Wlo,
        float* __restrict__ Bx) {
    const int lane  = threadIdx.x & 31;
    const int wave  = threadIdx.x >> 5;
    const int m0    = blockIdx.x * 32;
    const int e0    = (blockIdx.y * 4 + wave) * 64;
    const int l15   = lane & 15;
    const int khalf = lane >> 4;

    const size_t rowA0 = (size_t)(m0 + l15) * DDIM;       // M-tile 0 row
    const size_t rowA1 = rowA0 + (size_t)16 * DDIM;       // M-tile 1 row

    v8f acc[2][4];
#pragma unroll
    for (int m = 0; m < 2; ++m)
#pragma unroll
        for (int t = 0; t < 4; ++t) acc[m][t] = (v8f){};

    for (int k0 = 0; k0 < DDIM; k0 += 32) {
        // A fragment (16x32 bf16): lanes 0-15 K {0..7,16..23}, 16-31 K {8..15,24..31}
        const int ka = k0 + 8 * khalf;
        v16bf a_hi0 = cat8(*(const v8bf*)(Xhi + rowA0 + ka),
                           *(const v8bf*)(Xhi + rowA0 + ka + 16));
        v16bf a_lo0 = cat8(*(const v8bf*)(Xlo + rowA0 + ka),
                           *(const v8bf*)(Xlo + rowA0 + ka + 16));
        v16bf a_hi1 = cat8(*(const v8bf*)(Xhi + rowA1 + ka),
                           *(const v8bf*)(Xhi + rowA1 + ka + 16));
        v16bf a_lo1 = cat8(*(const v8bf*)(Xlo + rowA1 + ka),
                           *(const v8bf*)(Xlo + rowA1 + ka + 16));

        // B fragment (32x16 bf16): lanes 0-15 K {0..7,16..23}, 16-31 K {8..15,24..31}
        const int kb = k0 + 8 * khalf;
#pragma unroll
        for (int t = 0; t < 4; ++t) {
            const size_t rowB = (size_t)(e0 + 16 * t + l15) * DDIM + kb;
            v16bf b_hi = cat8(*(const v8bf*)(Whi + rowB),
                              *(const v8bf*)(Whi + rowB + 16));
            v16bf b_lo = cat8(*(const v8bf*)(Wlo + rowB),
                              *(const v8bf*)(Wlo + rowB + 16));

            v8f a0 = acc[0][t];
            a0 = __builtin_amdgcn_wmma_f32_16x16x32_bf16(false, a_hi0, false, b_hi,
                                                         (short)0, a0, false, false);
            a0 = __builtin_amdgcn_wmma_f32_16x16x32_bf16(false, a_hi0, false, b_lo,
                                                         (short)0, a0, false, false);
            a0 = __builtin_amdgcn_wmma_f32_16x16x32_bf16(false, a_lo0, false, b_hi,
                                                         (short)0, a0, false, false);
            acc[0][t] = a0;

            v8f a1 = acc[1][t];
            a1 = __builtin_amdgcn_wmma_f32_16x16x32_bf16(false, a_hi1, false, b_hi,
                                                         (short)0, a1, false, false);
            a1 = __builtin_amdgcn_wmma_f32_16x16x32_bf16(false, a_hi1, false, b_lo,
                                                         (short)0, a1, false, false);
            a1 = __builtin_amdgcn_wmma_f32_16x16x32_bf16(false, a_lo1, false, b_hi,
                                                         (short)0, a1, false, false);
            acc[1][t] = a1;
        }
    }

    // C/D layout: VGPR r -> (M = r + 8*khalf, N = lane&15)
#pragma unroll
    for (int m = 0; m < 2; ++m) {
#pragma unroll
        for (int r = 0; r < 8; ++r) {
            size_t rbase = (size_t)(m0 + 16 * m + r + 8 * khalf) * DDIM + e0 + l15;
            Bx[rbase +  0] = acc[m][0][r];
            Bx[rbase + 16] = acc[m][1][r];
            Bx[rbase + 32] = acc[m][2][r];
            Bx[rbase + 48] = acc[m][3][r];
        }
    }
}

// -------------------------------------------------------------------------
// Phase 1: per-chunk local scan from zero state; record chunk-final (u,v).
// Thread handles 4 consecutive d channels of one (b, chunk). 32768 threads.
// -------------------------------------------------------------------------
__global__ __launch_bounds__(256) void scan_phase1(
        const float* __restrict__ Bx,
        const float* __restrict__ omega_raw,
        float* __restrict__ FU, float* __restrict__ FV) {
    const int idx = blockIdx.x * blockDim.x + threadIdx.x;
    const int d4 = idx & 255;
    const int c  = (idx >> 8) & (CHUNKS - 1);
    const int b  = idx >> 13;
    const int d  = d4 * 4;

    float4 w4;
    w4.x = softplus_f(omega_raw[d + 0]);
    w4.y = softplus_f(omega_raw[d + 1]);
    w4.z = softplus_f(omega_raw[d + 2]);
    w4.w = softplus_f(omega_raw[d + 3]);
    const float h = DT_STEP;

    float4 u = {0.f, 0.f, 0.f, 0.f}, v = {0.f, 0.f, 0.f, 0.f};
    const size_t base = ((size_t)b * TSEQ + (size_t)c * CLEN) * DDIM + d;

    for (int t = 0; t < CLEN; ++t) {
        __builtin_prefetch((const void*)(Bx + base + (size_t)(t + 4) * DDIM), 0, 0);
        float4 f = *(const float4*)(Bx + base + (size_t)t * DDIM);
        u.x += h * (-w4.x * v.x + f.x); v.x += h * u.x;
        u.y += h * (-w4.y * v.y + f.y); v.y += h * u.y;
        u.z += h * (-w4.z * v.z + f.z); v.z += h * u.z;
        u.w += h * (-w4.w * v.w + f.w); v.w += h * u.w;
    }
    const size_t s = ((size_t)b * CHUNKS + c) * DDIM + d;
    *(float4*)(FU + s) = u;
    *(float4*)(FV + s) = v;
}

// -------------------------------------------------------------------------
// Phase 2: sequential combine of chunk summaries.  Per-step transition
// S = [[1, -h*w],[h, 1 - h^2*w]];  state_end = S^CLEN * state_start + f.
// S^128 via 7 repeated squarings.  4096 threads.
// -------------------------------------------------------------------------
__global__ __launch_bounds__(256) void scan_phase2(
        const float* __restrict__ omega_raw,
        const float* __restrict__ FU, const float* __restrict__ FV,
        float* __restrict__ SU, float* __restrict__ SV) {
    const int idx = blockIdx.x * blockDim.x + threadIdx.x;
    const int d = idx & (DDIM - 1);
    const int b = idx >> 10;
    const float w = softplus_f(omega_raw[d]);
    const float h = DT_STEP;

    float p00 = 1.f, p01 = -h * w, p10 = h, p11 = 1.f - h * h * w;
#pragma unroll
    for (int i = 0; i < 7; ++i) {        // S^(2^7) = S^CLEN
        float q00 = p00 * p00 + p01 * p10;
        float q01 = p00 * p01 + p01 * p11;
        float q10 = p10 * p00 + p11 * p10;
        float q11 = p10 * p01 + p11 * p11;
        p00 = q00; p01 = q01; p10 = q10; p11 = q11;
    }

    const size_t base = (size_t)b * CHUNKS * DDIM + d;
    float u = 0.f, v = 0.f;
    SU[base] = 0.f;
    SV[base] = 0.f;
    for (int c = 1; c < CHUNKS; ++c) {
        const size_t fi = base + (size_t)(c - 1) * DDIM;
        const float fu = FU[fi], fv = FV[fi];
        const float nu = p00 * u + p01 * v + fu;
        const float nv = p10 * u + p11 * v + fv;
        u = nu; v = nv;
        SU[base + (size_t)c * DDIM] = u;
        SV[base + (size_t)c * DDIM] = v;
    }
}

// -------------------------------------------------------------------------
// Phase 3: re-run each chunk from its corrected initial state, write
// y (= v, in place over the Bx staging region), u_seq, v_seq.
// -------------------------------------------------------------------------
__global__ __launch_bounds__(256) void scan_phase3(
        float* __restrict__ Y,           // holds Bx on entry; becomes y_seq
        float* __restrict__ U, float* __restrict__ V,
        const float* __restrict__ omega_raw,
        const float* __restrict__ SU, const float* __restrict__ SV) {
    const int idx = blockIdx.x * blockDim.x + threadIdx.x;
    const int d4 = idx & 255;
    const int c  = (idx >> 8) & (CHUNKS - 1);
    const int b  = idx >> 13;
    const int d  = d4 * 4;

    float4 w4;
    w4.x = softplus_f(omega_raw[d + 0]);
    w4.y = softplus_f(omega_raw[d + 1]);
    w4.z = softplus_f(omega_raw[d + 2]);
    w4.w = softplus_f(omega_raw[d + 3]);
    const float h = DT_STEP;

    const size_t s = ((size_t)b * CHUNKS + c) * DDIM + d;
    float4 u = *(const float4*)(SU + s);
    float4 v = *(const float4*)(SV + s);

    const size_t base = ((size_t)b * TSEQ + (size_t)c * CLEN) * DDIM + d;
    for (int t = 0; t < CLEN; ++t) {
        const size_t o = base + (size_t)t * DDIM;
        __builtin_prefetch((const void*)(Y + o + 4 * DDIM), 0, 0);
        float4 f = *(const float4*)(Y + o);
        u.x += h * (-w4.x * v.x + f.x); v.x += h * u.x;
        u.y += h * (-w4.y * v.y + f.y); v.y += h * u.y;
        u.z += h * (-w4.z * v.z + f.z); v.z += h * u.z;
        u.w += h * (-w4.w * v.w + f.w); v.w += h * u.w;
        *(float4*)(Y + o) = v;           // y_seq == v_seq (read-before-write, 1:1 ownership)
        *(float4*)(U + o) = u;
        *(float4*)(V + o) = v;
    }
}

extern "C" void kernel_launch(void* const* d_in, const int* in_sizes, int n_in,
                              void* d_out, int out_size, void* d_ws, size_t ws_size,
                              hipStream_t stream) {
    (void)in_sizes; (void)n_in; (void)out_size; (void)ws_size;
    const float* X         = (const float*)d_in[0];   // x_seq  [4,4096,1024]
    const float* omega_raw = (const float*)d_in[1];   // [1024]
    const float* W         = (const float*)d_in[2];   // B_weight [1024,1024]

    float* out = (float*)d_out;
    float* Y = out;                 // Bx staging -> y_seq
    float* U = out + NTOT;
    float* V = out + 2 * NTOT;

    // bf16 hi/lo staging: Xhi+Xlo exactly fill the U region (NTOT floats ==
    // 2*NTOT bf16); Whi+Wlo sit at the front of the V region.  Both are fully
    // consumed by the GEMM before phase 3 rewrites U and V.
    __bf16* Xhi = (__bf16*)U;
    __bf16* Xlo = Xhi + NTOT;
    __bf16* Whi = (__bf16*)V;
    __bf16* Wlo = Whi + WTOT;

    // Chunk-summary scratch: 4 * BSZ*CHUNKS*DDIM floats = 2 MB.
    const size_t SSZ = (size_t)BSZ * CHUNKS * DDIM;
    float* FU = (float*)d_ws;
    float* FV = FU + SSZ;
    float* SU = FV + SSZ;
    float* SV = SU + SSZ;

    convert_hilo<<<(int)(NTOT / 8 / 256), 256, 0, stream>>>(X, Xhi, Xlo);
    convert_hilo<<<(int)(WTOT / 8 / 256), 256, 0, stream>>>(W, Whi, Wlo);

    // GEMM: 512 M-pairs x 16 N-groups; 4 waves/block each take one N-group.
    dim3 ggrid(BSZ * TSEQ / 32, 4, 1);
    gemm_bx_kernel<<<ggrid, 128, 0, stream>>>(Xhi, Xlo, Whi, Wlo, Y);

    const int scan_threads = BSZ * CHUNKS * (DDIM / 4);  // 32768
    scan_phase1<<<scan_threads / 256, 256, 0, stream>>>(Y, omega_raw, FU, FV);
    scan_phase2<<<BSZ * DDIM / 256, 256, 0, stream>>>(omega_raw, FU, FV, SU, SV);
    scan_phase3<<<scan_threads / 256, 256, 0, stream>>>(Y, U, V, omega_raw, SU, SV);
}
